// ScaledDotProductAttention_16578573763340
// MI455X (gfx1250) — matrix operand order repaired and checked by reading, not compiled, not run on hardware
//
#include <hip/hip_runtime.h>

// ---------------------------------------------------------------------------
// Scaled dot-product attention for MI455X (gfx1250), wave32 + WMMA + TDM.
//   Q = X@Wq^T+bq ; K = X@Wk^T+bk ; V = X@Wv^T+bv
//   Out = softmax(Q K^T / sqrt(D)) @ V
// fp32 I/O, f16 operand storage, f32 WMMA accumulation.
// B operand of every GEMM is DMA'd into LDS by the Tensor Data Mover
// (double-buffered, TENSORcnt-synchronized), shared by all 8 waves of a block.
// Inner loop batches LDS fragment loads and pipelines the A operand one
// k-step ahead so waits overlap the WMMA pipe instead of serializing it.
// ---------------------------------------------------------------------------

typedef _Float16 v16h __attribute__((ext_vector_type(16)));
typedef _Float16 v8h  __attribute__((ext_vector_type(8)));
typedef _Float16 h4   __attribute__((ext_vector_type(4)));
typedef float    v8f  __attribute__((ext_vector_type(8)));
typedef unsigned int u32x4 __attribute__((ext_vector_type(4)));
typedef int          i32x4 __attribute__((ext_vector_type(4)));
typedef int          i32x8 __attribute__((ext_vector_type(8)));

constexpr int SEQ = 4096;
constexpr int DIM = 1024;
constexpr float SCALE = 0.03125f;     // 1/sqrt(1024)

constexpr int KC      = 128;          // K-chunk per LDS stage
constexpr int BSTRIDE = KC + 16;      // LDS halfs per column (32B pad -> bank spread)
constexpr int BSTAGE  = 64 * BSTRIDE; // halfs per stage buffer (18 KB)

#if defined(__has_builtin)
#if __has_builtin(__builtin_amdgcn_tensor_load_to_lds) && \
    __has_builtin(__builtin_amdgcn_s_wait_tensorcnt)
#define HAVE_TDM 1
#endif
#endif
#ifndef HAVE_TDM
#define HAVE_TDM 0
#endif

// ---- WMMA fragment loaders (contiguous-k order, same for A and B, wave32) -------------------
// A 16x32 f16: lane m(0-15): row m, halfs 0..15 = K0..15
//              lane m+16   : row m, halfs 0..15 = K16..31
__device__ __forceinline__ v16h load_a_frag(const _Float16* __restrict__ A,
                                            int lda, int row, int k0, int lane) {
  const _Float16* p = A + (size_t)row * lda + k0 + ((lane >> 4) << 4);
  v8h lo = *(const v8h*)p;
  v8h hi = *(const v8h*)(p + 8);
  return __builtin_shufflevector(lo, hi, 0,1,2,3,4,5,6,7,8,9,10,11,12,13,14,15);
}

// B 32x16 fragment from the LDS panel: lane n(0-15) = column n, K k0..k0+15;
// lanes 16-31 = same column, K k0+16..k0+31 (contiguous 32B per lane).
__device__ __forceinline__ v16h load_b_lds(const _Float16* cur, int cl, int bko,
                                           int ks, int j) {
  return *(const v16h*)(cur + (j * 16 + cl) * BSTRIDE + ks + bko);
}

// ---- TDM: DMA one 64-column x KC-half B panel (row stride Kdim) into LDS --
// D# pad feature inserts 8 DWORDs (32B) after every 64 DWORDs (256B = one
// column row), producing the padded BSTRIDE layout directly.
__device__ __forceinline__ void tdm_issue_b_panel(const _Float16* gsrc,
                                                  unsigned lds_addr, int Kdim) {
#if HAVE_TDM
  unsigned long long ga = (unsigned long long)(size_t)gsrc;
  u32x4 g0;
  g0[0] = 1u;                                              // count=1, user D#
  g0[1] = lds_addr;                                        // lds_addr (bytes)
  g0[2] = (unsigned)(ga & 0xFFFFFFFFu);                    // global_addr lo
  g0[3] = (unsigned)((ga >> 32) & 0x1FFFFFFu) | (2u << 30);// global hi | type=2
  i32x8 g1;
  g1[0] = (1 << 16)        // data_size = 1 -> 2 bytes
        | (1 << 20)        // pad_enable
        | (5 << 22)        // pad_interval = 5 -> every 64 DWORDs (256B)
        | (7 << 25);       // pad_amount   = 7 -> 8 DWORDs (32B)
  g1[1] = (KC & 0xFFFF) << 16;        // tensor_dim0[15:0]
  g1[2] = (64 << 16);                 // tensor_dim0 hi=0 | tensor_dim1[15:0]=64
  g1[3] = (KC << 16);                 // tensor_dim1 hi=0 | tile_dim0=KC
  g1[4] = 64;                         // tile_dim1=64 | tile_dim2=0
  g1[5] = Kdim;                       // tensor_dim0_stride lo (elements)
  g1[6] = 0;                          // stride0 hi | stride1 lo
  g1[7] = 0;
  i32x4 z4 = {0, 0, 0, 0};
#if __clang_major__ >= 23
  i32x8 z8 = {0, 0, 0, 0, 0, 0, 0, 0};
  __builtin_amdgcn_tensor_load_to_lds(g0, g1, z4, z4, z8, 0);
#else
  __builtin_amdgcn_tensor_load_to_lds(g0, g1, z4, z4, 0);
#endif
#else
  (void)gsrc; (void)lds_addr; (void)Kdim;
#endif
}

// Issue the fetch of one B stage (wave 0 via TDM, or cooperative fallback).
__device__ __forceinline__ void stage_fetch(const _Float16* __restrict__ Bt,
                                            int Kdim, int n0, int kb,
                                            _Float16* buf) {
#if HAVE_TDM
  if (threadIdx.x < 32) {  // wave 0 only; EXEC ignored by tensor ops
    tdm_issue_b_panel(Bt + (size_t)n0 * Kdim + kb, (unsigned)(size_t)buf, Kdim);
  }
#else
  const int tid = threadIdx.x;
  const int col = tid >> 2;            // 64 columns, 4 threads each
#pragma unroll
  for (int i = 0; i < 4; ++i) {
    const int chunk = (tid & 3) + i * 4;  // 16 chunks of 8 halfs per column
    *(v8h*)(buf + col * BSTRIDE + chunk * 8) =
        *(const v8h*)(Bt + (size_t)(n0 + col) * Kdim + kb + chunk * 8);
  }
#endif
}

// Wait for the current stage's DMA, then publish to all waves.
__device__ __forceinline__ void stage_wait(bool next_in_flight) {
#if HAVE_TDM
  if (threadIdx.x < 32) {
    if (next_in_flight) __builtin_amdgcn_s_wait_tensorcnt((short)1);
    else                __builtin_amdgcn_s_wait_tensorcnt((short)0);
  }
#else
  (void)next_in_flight;
#endif
  __syncthreads();
}

// One wave computes a 16(M) x 64(N) strip; B comes from the shared LDS panel.
// Block = 8 waves -> 128x64 output tile; B panel double-buffered via TDM.
__device__ __forceinline__ void gemm_nt_staged(const _Float16* __restrict__ A,
                                               const _Float16* __restrict__ Bt,
                                               int Kdim, int m0, int n0,
                                               _Float16* bs, v8f acc[4]) {
  const int lane = threadIdx.x & 31;
  const int arow = m0 + (lane & 15);
  const int cl   = lane & 15;
  const int bko  = (lane >> 4) << 4;   // +0 / +16 halfs (B layout, upper half-wave)
  const _Float16* aprefetch = A + (size_t)arow * Kdim;

  stage_fetch(Bt, Kdim, n0, 0, bs);
  const int nstage = Kdim / KC;
  for (int s = 0; s < nstage; ++s) {
    _Float16* cur = bs + (s & 1) * BSTAGE;
    _Float16* nxt = bs + ((s + 1) & 1) * BSTAGE;
    const int kb = s * KC;
    if (s + 1 < nstage) {
      stage_fetch(Bt, Kdim, n0, kb + KC, nxt);  // nxt free: barrier at prev tail
      stage_wait(true);
    } else {
      stage_wait(false);
    }
    __builtin_prefetch(aprefetch + kb + 2 * KC, 0, 0);  // A stream -> L2

    // Pipelined MMA: A fragment fetched one k-step ahead; all 4 B fragments
    // issued before the WMMA quad so LDS waits overlap the matrix pipe.
    v16h a = load_a_frag(A, Kdim, arow, kb, lane);
#pragma unroll
    for (int ks = 0; ks < KC; ks += 32) {
      v16h b0 = load_b_lds(cur, cl, bko, ks, 0);
      v16h b1 = load_b_lds(cur, cl, bko, ks, 1);
      v16h b2 = load_b_lds(cur, cl, bko, ks, 2);
      v16h b3 = load_b_lds(cur, cl, bko, ks, 3);
      v16h an = a;
      if (ks + 32 < KC) an = load_a_frag(A, Kdim, arow, kb + ks + 32, lane);
      acc[0] = __builtin_amdgcn_wmma_f32_16x16x32_f16(false, a, false, b0,
                                                      (short)0, acc[0], false, false);
      acc[1] = __builtin_amdgcn_wmma_f32_16x16x32_f16(false, a, false, b1,
                                                      (short)0, acc[1], false, false);
      acc[2] = __builtin_amdgcn_wmma_f32_16x16x32_f16(false, a, false, b2,
                                                      (short)0, acc[2], false, false);
      acc[3] = __builtin_amdgcn_wmma_f32_16x16x32_f16(false, a, false, b3,
                                                      (short)0, acc[3], false, false);
      a = an;
    }
    __syncthreads();  // all waves done reading cur -> reusable as nxt
  }
}

// ---- phase 0: fp32 -> fp16 conversion -------------------------------------
__global__ __launch_bounds__(256) void f32_to_f16_kernel(
    const float* __restrict__ src, _Float16* __restrict__ dst, int n) {
  int i = (blockIdx.x * 256 + threadIdx.x) * 4;
  if (i < n) {
    float4 v = *(const float4*)(src + i);
    h4 o = {(_Float16)v.x, (_Float16)v.y, (_Float16)v.z, (_Float16)v.w};
    *(h4*)(dst + i) = o;
  }
}

// ---- phase 1: QKV projection (out = X @ W^T + b), optional transposed store
__global__ __launch_bounds__(256) void qkv_gemm_kernel(
    const _Float16* __restrict__ Xh, const _Float16* __restrict__ Wh,
    const float* __restrict__ bias, _Float16* __restrict__ out, int transposed) {
  __shared__ __align__(32) _Float16 Bs[2 * BSTAGE];
  const int lane = threadIdx.x & 31;
  const int wave = threadIdx.x >> 5;
  const int m0 = blockIdx.x * 128 + wave * 16;
  const int n0 = blockIdx.y * 64;
  v8f acc[4] = {};
  gemm_nt_staged(Xh, Wh, DIM, m0, n0, Bs, acc);
  const int rbase = (lane >> 4) << 3;
  const int cl = lane & 15;
#pragma unroll
  for (int j = 0; j < 4; ++j) {
    const int col = n0 + j * 16 + cl;
    const float b = bias[col];
#pragma unroll
    for (int r = 0; r < 8; ++r) {
      const int row = m0 + rbase + r;
      const _Float16 v = (_Float16)(acc[j][r] + b);
      if (transposed) out[(size_t)col * SEQ + row] = v;   // Vt[d, s]
      else            out[(size_t)row * DIM + col] = v;   // Q/K[s, d]
    }
  }
}

// ---- phase 2: scores = (Q K^T) * scale, f32 -------------------------------
__global__ __launch_bounds__(256) void scores_gemm_kernel(
    const _Float16* __restrict__ Qh, const _Float16* __restrict__ Kh,
    float* __restrict__ Sc) {
  __shared__ __align__(32) _Float16 Bs[2 * BSTAGE];
  const int lane = threadIdx.x & 31;
  const int wave = threadIdx.x >> 5;
  const int m0 = blockIdx.x * 128 + wave * 16;
  const int n0 = blockIdx.y * 64;
  v8f acc[4] = {};
  gemm_nt_staged(Qh, Kh, DIM, m0, n0, Bs, acc);
  const int rbase = (lane >> 4) << 3;
  const int cl = lane & 15;
#pragma unroll
  for (int j = 0; j < 4; ++j) {
    const int col = n0 + j * 16 + cl;
#pragma unroll
    for (int r = 0; r < 8; ++r) {
      const int row = m0 + rbase + r;
      Sc[(size_t)row * SEQ + col] = acc[j][r] * SCALE;
    }
  }
}

// ---- phase 3: row softmax, f32 scores -> f16 probabilities ----------------
__global__ __launch_bounds__(256) void softmax_kernel(
    const float* __restrict__ Sc, _Float16* __restrict__ P) {
  __shared__ float red[256];
  const int row = blockIdx.x;
  const int tid = threadIdx.x;
  const float4* src = (const float4*)(Sc + (size_t)row * SEQ);
  float4 v[4];
  float mx = -3.402823466e38f;
#pragma unroll
  for (int i = 0; i < 4; ++i) {
    v[i] = src[tid + i * 256];
    mx = fmaxf(mx, fmaxf(fmaxf(v[i].x, v[i].y), fmaxf(v[i].z, v[i].w)));
  }
  red[tid] = mx;
  __syncthreads();
  for (int s = 128; s > 0; s >>= 1) {
    if (tid < s) red[tid] = fmaxf(red[tid], red[tid + s]);
    __syncthreads();
  }
  mx = red[0];
  __syncthreads();
  float sum = 0.f;
#pragma unroll
  for (int i = 0; i < 4; ++i) {
    v[i].x = __expf(v[i].x - mx);
    v[i].y = __expf(v[i].y - mx);
    v[i].z = __expf(v[i].z - mx);
    v[i].w = __expf(v[i].w - mx);
    sum += v[i].x + v[i].y + v[i].z + v[i].w;
  }
  red[tid] = sum;
  __syncthreads();
  for (int s = 128; s > 0; s >>= 1) {
    if (tid < s) red[tid] += red[tid + s];
    __syncthreads();
  }
  const float inv = 1.0f / red[0];
  h4* dst = (h4*)(P + (size_t)row * SEQ);
#pragma unroll
  for (int i = 0; i < 4; ++i) {
    h4 o = {(_Float16)(v[i].x * inv), (_Float16)(v[i].y * inv),
            (_Float16)(v[i].z * inv), (_Float16)(v[i].w * inv)};
    dst[tid + i * 256] = o;
  }
}

// ---- phase 4: Out = P @ V  (B operand = Vt rows, NT form), f32 out --------
__global__ __launch_bounds__(256) void out_gemm_kernel(
    const _Float16* __restrict__ P, const _Float16* __restrict__ Vth,
    float* __restrict__ Out) {
  __shared__ __align__(32) _Float16 Bs[2 * BSTAGE];
  const int lane = threadIdx.x & 31;
  const int wave = threadIdx.x >> 5;
  const int m0 = blockIdx.x * 128 + wave * 16;
  const int n0 = blockIdx.y * 64;
  v8f acc[4] = {};
  gemm_nt_staged(P, Vth, SEQ, m0, n0, Bs, acc);   // K = 4096
  const int rbase = (lane >> 4) << 3;
  const int cl = lane & 15;
#pragma unroll
  for (int j = 0; j < 4; ++j) {
    const int col = n0 + j * 16 + cl;
#pragma unroll
    for (int r = 0; r < 8; ++r) {
      const int row = m0 + rbase + r;
      Out[(size_t)row * DIM + col] = acc[j][r];
    }
  }
}

// ---------------------------------------------------------------------------
extern "C" void kernel_launch(void* const* d_in, const int* in_sizes, int n_in,
                              void* d_out, int out_size, void* d_ws, size_t ws_size,
                              hipStream_t stream) {
  const float* X  = (const float*)d_in[0];
  const float* Wq = (const float*)d_in[1];
  const float* bq = (const float*)d_in[2];
  const float* Wk = (const float*)d_in[3];
  const float* bk = (const float*)d_in[4];
  const float* Wv = (const float*)d_in[5];
  const float* bv = (const float*)d_in[6];
  float* Out = (float*)d_out;

  // Workspace layout (~134 MB): every buffer fully overwritten before read.
  char* ws = (char*)d_ws;
  size_t off = 0;
  auto take = [&](size_t bytes) { char* p = ws + off; off += bytes; return p; };
  _Float16* Xh  = (_Float16*)take((size_t)SEQ * DIM * 2);   //  8 MB
  _Float16* Wqh = (_Float16*)take((size_t)DIM * DIM * 2);   //  2 MB
  _Float16* Wkh = (_Float16*)take((size_t)DIM * DIM * 2);   //  2 MB
  _Float16* Wvh = (_Float16*)take((size_t)DIM * DIM * 2);   //  2 MB
  _Float16* Qh  = (_Float16*)take((size_t)SEQ * DIM * 2);   //  8 MB
  _Float16* Kh  = (_Float16*)take((size_t)SEQ * DIM * 2);   //  8 MB
  _Float16* Vth = (_Float16*)take((size_t)DIM * SEQ * 2);   //  8 MB (V^T)
  float*    Sc  = (float*)   take((size_t)SEQ * SEQ * 4);   // 64 MB (lives in L2)
  _Float16* P   = (_Float16*)take((size_t)SEQ * SEQ * 2);   // 32 MB
  (void)ws_size; (void)in_sizes; (void)n_in; (void)out_size;

  // phase 0: f32 -> f16 conversions
  f32_to_f16_kernel<<<(SEQ * DIM) / 1024, 256, 0, stream>>>(X,  Xh,  SEQ * DIM);
  f32_to_f16_kernel<<<(DIM * DIM) / 1024, 256, 0, stream>>>(Wq, Wqh, DIM * DIM);
  f32_to_f16_kernel<<<(DIM * DIM) / 1024, 256, 0, stream>>>(Wk, Wkh, DIM * DIM);
  f32_to_f16_kernel<<<(DIM * DIM) / 1024, 256, 0, stream>>>(Wv, Wvh, DIM * DIM);

  // phase 1: QKV projections (V stored transposed for the P@V GEMM)
  dim3 gQKV(SEQ / 128, DIM / 64);
  qkv_gemm_kernel<<<gQKV, 256, 0, stream>>>(Xh, Wqh, bq, Qh, 0);
  qkv_gemm_kernel<<<gQKV, 256, 0, stream>>>(Xh, Wkh, bk, Kh, 0);
  qkv_gemm_kernel<<<gQKV, 256, 0, stream>>>(Xh, Wvh, bv, Vth, 1);

  // phase 2: scores = Q K^T / sqrt(D)
  dim3 gS(SEQ / 128, SEQ / 64);
  scores_gemm_kernel<<<gS, 256, 0, stream>>>(Qh, Kh, Sc);

  // phase 3: row softmax -> f16 probabilities
  softmax_kernel<<<SEQ, 256, 0, stream>>>(Sc, P);

  // phase 4: Out = P @ V
  dim3 gO(SEQ / 128, DIM / 64);
  out_gemm_kernel<<<gO, 256, 0, stream>>>(P, Vth, Out);
}
